// GPTLanguageModel_16870631539283
// MI455X (gfx1250) — matrix operand order repaired and checked by reading, not compiled, not run on hardware
//
#include <hip/hip_runtime.h>

// ---------------------------------------------------------------------------
// GPT forward (4 layers) for MI455X / gfx1250, wave32, bf16 WMMA (f32 accum).
// All GEMMs go through one LDS-tiled kernel using v_wmma_f32_16x16x32_bf16.
// Block tile 256x64x32, 8 waves, each wave computes a 32x64 subtile (2x4 WMMA
// accumulators). Ping-pong LDS double buffering, one barrier per K-chunk:
//   frag ds_loads (buf p) -> issue next global_load_b128 -> 8x WMMA
//   -> ds_store staged tile (buf 1-p) -> s_barrier.
// ---------------------------------------------------------------------------

#define VV   32000
#define CC   768
#define TT   1024
#define BBA  2
#define HHD  12
#define DHD  64
#define LLY  4
#define HIDD 3072
#define BT   (BBA * TT)

typedef unsigned short u16;
typedef __attribute__((ext_vector_type(16))) __bf16 bf16x16;
typedef __attribute__((ext_vector_type(8)))  __bf16 bf16x8;
typedef __attribute__((ext_vector_type(8)))  float  fx8;

#define GF_RELU    1
#define GF_OUTBF16 2
#define GF_TRANS   4

__device__ __forceinline__ u16 f2bf(float f) {
  unsigned u = __float_as_uint(f);
  u += 0x7FFFu + ((u >> 16) & 1u);   // round-to-nearest-even
  return (u16)(u >> 16);
}

__device__ __forceinline__ float block_reduce_sum(float v, float* red) {
  int t = threadIdx.x;
  red[t] = v; __syncthreads();
  for (int off = 128; off > 0; off >>= 1) {
    if (t < off) red[t] += red[t + off];
    __syncthreads();
  }
  float r = red[0]; __syncthreads();
  return r;
}

__device__ __forceinline__ float block_reduce_max(float v, float* red) {
  int t = threadIdx.x;
  red[t] = v; __syncthreads();
  for (int off = 128; off > 0; off >>= 1) {
    if (t < off) red[t] = fmaxf(red[t], red[t + off]);
    __syncthreads();
  }
  float r = red[0]; __syncthreads();
  return r;
}

// ---------------------------------------------------------------------------
// Generic bf16 GEMM: C[m,n] = act( scale * sum_k A[m,k]*Bt[n,k] + bias + res )
// A:[M,K] row-major bf16 (lda), Bt:[N,K] row-major bf16 (ldb, i.e. B col-major)
// Requires: M % 256 == 0, N % 64 == 0, K % 32 == 0 (true for all uses here).
// ---------------------------------------------------------------------------
__launch_bounds__(256, 1)
__global__ void gemm_bf16_wmma_kernel(
    const u16* __restrict__ A, const u16* __restrict__ Bt, void* __restrict__ Cout,
    const float* __restrict__ bias, const float* __restrict__ resid,
    int K, int lda, int ldb, int ldc,
    long sAb, long sAh, long sBb, long sBh, long sCb, long sCh,
    int batchH, float scale, int flags)
{
  __shared__ __attribute__((aligned(16))) u16 As[2][256][40];  // 40 KB
  __shared__ __attribute__((aligned(16))) u16 Bs[2][64][40];   // 10 KB

  const int tid  = threadIdx.x;
  const int lane = tid & 31;
  const int wid  = tid >> 5;       // 0..7: wave owns rows [wid*32, wid*32+32)
  const int zb   = blockIdx.z / batchH;
  const int zh   = blockIdx.z % batchH;
  const u16* Ab  = A  + zb * sAb + zh * sAh;
  const u16* Bb  = Bt + zb * sBb + zh * sBh;
  const long coff = zb * sCb + zh * sCh;
  const int bm   = blockIdx.y * 256;
  const int bn   = blockIdx.x * 64;
  const int half = lane >> 4;      // lane half selects K sub-range
  const int l16  = lane & 15;

  // each thread moves 8 bf16 per 64-row tile-slab via b128
  const int lin = tid * 8;
  const int r0  = lin >> 5;        // 0..63
  const int c0  = lin & 31;
  const u16* gA0 = Ab + (long)(bm + r0)       * lda + c0;
  const u16* gA1 = Ab + (long)(bm + r0 + 64)  * lda + c0;
  const u16* gA2 = Ab + (long)(bm + r0 + 128) * lda + c0;
  const u16* gA3 = Ab + (long)(bm + r0 + 192) * lda + c0;
  const u16* gB0 = Bb + (long)(bn + r0)       * ldb + c0;

  fx8 acc[2][4] = {};

  // ---- prologue: fetch K-chunk 0 and place it in buffer 0 ----
  {
    uint4 a0 = *(const uint4*)gA0;
    uint4 a1 = *(const uint4*)gA1;
    uint4 a2 = *(const uint4*)gA2;
    uint4 a3 = *(const uint4*)gA3;
    uint4 b0 = *(const uint4*)gB0;
    *(uint4*)&As[0][r0][c0]       = a0;
    *(uint4*)&As[0][r0 + 64][c0]  = a1;
    *(uint4*)&As[0][r0 + 128][c0] = a2;
    *(uint4*)&As[0][r0 + 192][c0] = a3;
    *(uint4*)&Bs[0][r0][c0]       = b0;
  }
  __syncthreads();

  int p = 0;
  for (int k0 = 0; k0 < K; k0 += 32) {
    // ---- LDS -> fragments from buffer p (per ISA 7.12.2 layouts) ----
    bf16x16 af[2], bfr[4];
#pragma unroll
    for (int i = 0; i < 2; ++i) {
      // A 16x32 bf16: lanes 0-15 = rows, elems 0-7 -> K=base..base+7,
      // elems 8-15 -> K=base+16..base+23, base = 8*(lane/16)
      const u16* rowp = &As[p][wid * 32 + i * 16 + l16][half * 8];
      bf16x8 lo = *(const bf16x8*)(rowp);
      bf16x8 hi = *(const bf16x8*)(rowp + 16);
      af[i] = __builtin_shufflevector(lo, hi, 0,1,2,3,4,5,6,7,8,9,10,11,12,13,14,15);
    }
#pragma unroll
    for (int j = 0; j < 4; ++j) {
      // B 32x16 bf16: lane -> column, elems 0-7 -> K=base..base+7,
      // elems 8-15 -> K=base+16..base+23, base = 8*(lane/16)
      const u16* rowp = &Bs[p][j * 16 + l16][half * 8];
      bf16x8 lo = *(const bf16x8*)(rowp);
      bf16x8 hi = *(const bf16x8*)(rowp + 16);
      bfr[j] = __builtin_shufflevector(lo, hi, 0,1,2,3,4,5,6,7,8,9,10,11,12,13,14,15);
    }

    const bool more = (k0 + 32) < K;
    uint4 a0, a1, a2, a3, b0;
    if (more) {
      // issue next tile's global loads; latency hides under the WMMAs below
      int kn = k0 + 32;
      a0 = *(const uint4*)(gA0 + kn);
      a1 = *(const uint4*)(gA1 + kn);
      a2 = *(const uint4*)(gA2 + kn);
      a3 = *(const uint4*)(gA3 + kn);
      b0 = *(const uint4*)(gB0 + kn);
    }

#pragma unroll
    for (int i = 0; i < 2; ++i)
#pragma unroll
      for (int j = 0; j < 4; ++j)
        acc[i][j] = __builtin_amdgcn_wmma_f32_16x16x32_bf16(
            false, af[i], false, bfr[j], (short)0, acc[i][j], false, false);

    if (more) {
      int q = p ^ 1;
      *(uint4*)&As[q][r0][c0]       = a0;
      *(uint4*)&As[q][r0 + 64][c0]  = a1;
      *(uint4*)&As[q][r0 + 128][c0] = a2;
      *(uint4*)&As[q][r0 + 192][c0] = a3;
      *(uint4*)&Bs[q][r0][c0]       = b0;
    }
    __syncthreads();
    p ^= 1;
  }

  // ---- epilogue: C/D f32 layout: VGPR r -> M = r + 8*(lane/16), N = lane%16 ----
#pragma unroll
  for (int i = 0; i < 2; ++i) {
#pragma unroll
    for (int j = 0; j < 4; ++j) {
#pragma unroll
      for (int r = 0; r < 8; ++r) {
        int m = bm + wid * 32 + i * 16 + half * 8 + r;
        int n = bn + j * 16 + l16;
        float v = acc[i][j][r] * scale;
        if (bias)  v += bias[n];
        if (resid) v += resid[coff + (long)m * ldc + n];
        if (flags & GF_RELU) v = fmaxf(v, 0.0f);
        long cidx = coff + ((flags & GF_TRANS) ? ((long)n * ldc + m)
                                               : ((long)m * ldc + n));
        if (flags & GF_OUTBF16) ((u16*)Cout)[cidx]   = f2bf(v);
        else                    ((float*)Cout)[cidx] = v;
      }
    }
  }
}

// ---------------------------------------------------------------------------
// Small helper kernels
// ---------------------------------------------------------------------------
__global__ void embed_kernel(const int* __restrict__ idx, const float* __restrict__ tok,
                             const float* __restrict__ pos, float* __restrict__ x)
{
  long i = (long)blockIdx.x * 256 + threadIdx.x;   // over BT*CC
  int c = (int)(i % CC);
  long row = i / CC;
  int t = (int)(row % TT);
  x[i] = tok[(long)idx[row] * CC + c] + pos[(long)t * CC + c];
}

// out[(l*C + n)*C + k] = W[l, n/64, k, n%64]   (Wq/Wk/Wv: [L,H,C,DH] -> [L,N,K])
__global__ void pack_qkv_kernel(const float* __restrict__ W, u16* __restrict__ out)
{
  long i = (long)blockIdx.x * 256 + threadIdx.x;   // over L*C*C
  int k = (int)(i % CC);
  long t = i / CC;
  int n = (int)(t % CC);
  int l = (int)(t / CC);
  int h = n >> 6, d = n & 63;
  out[i] = f2bf(W[(((long)l * HHD + h) * CC + k) * DHD + d]);
}

// out[(b*Nd + n)*Kd + k] = in[(b*Kd + k)*Nd + n]  (transpose-convert, batched)
__global__ void pack_t_kernel(const float* __restrict__ in, u16* __restrict__ out,
                              int Kd, int Nd)
{
  long i = (long)blockIdx.x * 256 + threadIdx.x;
  int k = (int)(i % Kd);
  long t = i / Kd;
  int n = (int)(t % Nd);
  long b = t / Nd;
  out[i] = f2bf(in[(b * Kd + k) * (long)Nd + n]);
}

__global__ void f32_to_bf16_kernel(const float* __restrict__ in, u16* __restrict__ out)
{
  long i = (long)blockIdx.x * 256 + threadIdx.x;
  out[i] = f2bf(in[i]);
}

__global__ void layernorm_kernel(const float* __restrict__ x, const float* __restrict__ g,
                                 const float* __restrict__ b, u16* __restrict__ out)
{
  __shared__ float red[256];
  long row = blockIdx.x;
  const float* xr = x + row * CC;
  float s = 0.f;
  for (int c = threadIdx.x; c < CC; c += 256) s += xr[c];
  float mu = block_reduce_sum(s, red) * (1.0f / CC);
  float v = 0.f;
  for (int c = threadIdx.x; c < CC; c += 256) { float d = xr[c] - mu; v += d * d; }
  float var = block_reduce_sum(v, red) * (1.0f / CC);
  float inv = rsqrtf(var + 1e-5f);
  u16* orow = out + row * CC;
  for (int c = threadIdx.x; c < CC; c += 256)
    orow[c] = f2bf((xr[c] - mu) * inv * g[c] + b[c]);
}

// One block per (b,h,t) row of scores; causal softmax; fp32 in, bf16 out.
__global__ void softmax_causal_kernel(const float* __restrict__ S, u16* __restrict__ P)
{
  __shared__ float red[256];
  long row = blockIdx.x;             // (b*H + h)*T + t
  int t = (int)(row % TT);
  const float* sr = S + row * (long)TT;
  u16* pr = P + row * (long)TT;
  float mx = -3.402823466e38f;
  for (int s = threadIdx.x; s <= t; s += 256) mx = fmaxf(mx, sr[s]);
  mx = block_reduce_max(mx, red);
  float sum = 0.f;
  for (int s = threadIdx.x; s <= t; s += 256) sum += expf(sr[s] - mx);
  sum = block_reduce_sum(sum, red);
  float inv = 1.0f / sum;
  for (int s = threadIdx.x; s < TT; s += 256)
    pr[s] = (s <= t) ? f2bf(expf(sr[s] - mx) * inv) : (u16)0;
}

__global__ void row_loss_kernel(const float* __restrict__ logits, const int* __restrict__ tgt,
                                float* __restrict__ rowloss)
{
  __shared__ float red[256];
  long row = blockIdx.x;
  const float* lr = logits + row * (long)VV;
  float mx = -3.402823466e38f;
  for (int i = threadIdx.x; i < VV; i += 256) mx = fmaxf(mx, lr[i]);
  mx = block_reduce_max(mx, red);
  float s = 0.f;
  for (int i = threadIdx.x; i < VV; i += 256) s += expf(lr[i] - mx);
  s = block_reduce_sum(s, red);
  if (threadIdx.x == 0)
    rowloss[row] = -(lr[tgt[row]] - mx - logf(s));
}

__global__ void loss_reduce_kernel(const float* __restrict__ rowloss, float* __restrict__ out)
{
  __shared__ float red[256];
  float s = 0.f;
  for (int i = threadIdx.x; i < BT; i += 256) s += rowloss[i];
  s = block_reduce_sum(s, red);
  if (threadIdx.x == 0) *out = s * (1.0f / BT);
}

// ---------------------------------------------------------------------------
// Launcher
// ---------------------------------------------------------------------------
extern "C" void kernel_launch(void* const* d_in, const int* in_sizes, int n_in,
                              void* d_out, int out_size, void* d_ws, size_t ws_size,
                              hipStream_t stream)
{
  (void)in_sizes; (void)n_in; (void)out_size; (void)ws_size;

  const int*   idx     = (const int*)  d_in[0];
  const int*   targets = (const int*)  d_in[1];
  const float* tok_emb = (const float*)d_in[2];
  const float* pos_emb = (const float*)d_in[3];
  const float* Wq      = (const float*)d_in[4];
  const float* Wk      = (const float*)d_in[5];
  const float* Wv      = (const float*)d_in[6];
  const float* Wproj   = (const float*)d_in[7];
  const float* bproj   = (const float*)d_in[8];
  const float* ln1_g   = (const float*)d_in[9];
  const float* ln1_b   = (const float*)d_in[10];
  const float* ln2_g   = (const float*)d_in[11];
  const float* ln2_b   = (const float*)d_in[12];
  const float* W1      = (const float*)d_in[13];
  const float* b1      = (const float*)d_in[14];
  const float* W2      = (const float*)d_in[15];
  const float* b2      = (const float*)d_in[16];
  const float* Wlm     = (const float*)d_in[17];
  const float* blm     = (const float*)d_in[18];
  float* out = (float*)d_out;                        // logits [2048, 32000] + loss

  // ---- workspace carve-out (~295 MB total) ----
  char* wsb = (char*)d_ws;
  size_t off = 0;
  auto alloc = [&](size_t bytes) -> char* {
    char* p = wsb + off;
    off = (off + bytes + 255) & ~(size_t)255;
    return p;
  };
  u16*   WQT  = (u16*)alloc((size_t)LLY * CC * CC * 2);
  u16*   WKT  = (u16*)alloc((size_t)LLY * CC * CC * 2);
  u16*   WVT  = (u16*)alloc((size_t)LLY * CC * CC * 2);
  u16*   WPT  = (u16*)alloc((size_t)LLY * CC * CC * 2);
  u16*   W1T  = (u16*)alloc((size_t)LLY * CC * HIDD * 2);
  u16*   W2T  = (u16*)alloc((size_t)LLY * HIDD * CC * 2);
  u16*   WLMT = (u16*)alloc((size_t)CC * VV * 2);
  float* XF   = (float*)alloc((size_t)BT * CC * 4);
  u16*   HBF  = (u16*)alloc((size_t)BT * CC * 2);
  u16*   QBF  = (u16*)alloc((size_t)BT * CC * 2);
  u16*   KBF  = (u16*)alloc((size_t)BT * CC * 2);
  u16*   VTB  = (u16*)alloc((size_t)BT * CC * 2);        // [b][h*64+d][t]
  u16*   OBF  = (u16*)alloc((size_t)BT * CC * 2);
  u16*   A1   = (u16*)alloc((size_t)BT * HIDD * 2);
  u16*   XBF  = (u16*)alloc((size_t)BT * CC * 2);
  float* SC   = (float*)alloc((size_t)BBA * HHD * TT * TT * 4);
  u16*   PR   = (u16*)alloc((size_t)BBA * HHD * TT * TT * 2);
  float* RL   = (float*)alloc((size_t)BT * 4);

  auto gemm = [&](const u16* A, const u16* Bt, void* C, const float* bias,
                  const float* resid, int M, int N, int K, int lda, int ldb, int ldc,
                  long sAb, long sAh, long sBb, long sBh, long sCb, long sCh,
                  int bB, int bH, float scale, int flags) {
    dim3 grid(N / 64, M / 256, bB * bH);
    gemm_bf16_wmma_kernel<<<grid, 256, 0, stream>>>(
        A, Bt, C, bias, resid, K, lda, ldb, ldc,
        sAb, sAh, sBb, sBh, sCb, sCh, bH, scale, flags);
  };

  // ---- 1) pack weights fp32 -> bf16 transposed ([N,K], K contiguous) ----
  pack_qkv_kernel<<<(LLY * CC * CC) / 256, 256, 0, stream>>>(Wq, WQT);
  pack_qkv_kernel<<<(LLY * CC * CC) / 256, 256, 0, stream>>>(Wk, WKT);
  pack_qkv_kernel<<<(LLY * CC * CC) / 256, 256, 0, stream>>>(Wv, WVT);
  pack_t_kernel<<<(LLY * CC * CC) / 256, 256, 0, stream>>>(Wproj, WPT, CC, CC);
  pack_t_kernel<<<(LLY * CC * HIDD) / 256, 256, 0, stream>>>(W1, W1T, CC, HIDD);
  pack_t_kernel<<<(LLY * HIDD * CC) / 256, 256, 0, stream>>>(W2, W2T, HIDD, CC);
  pack_t_kernel<<<((size_t)CC * VV) / 256, 256, 0, stream>>>(Wlm, WLMT, CC, VV);

  // ---- 2) embeddings ----
  embed_kernel<<<(BT * CC) / 256, 256, 0, stream>>>(idx, tok_emb, pos_emb, XF);

  // ---- 3) transformer layers ----
  for (int l = 0; l < LLY; ++l) {
    const u16* wq = WQT + (size_t)l * CC * CC;
    const u16* wk = WKT + (size_t)l * CC * CC;
    const u16* wv = WVT + (size_t)l * CC * CC;
    const u16* wp = WPT + (size_t)l * CC * CC;
    const u16* w1 = W1T + (size_t)l * CC * HIDD;
    const u16* w2 = W2T + (size_t)l * HIDD * CC;

    layernorm_kernel<<<BT, 256, 0, stream>>>(XF, ln1_g + l * CC, ln1_b + l * CC, HBF);

    // q,k : [2048,768] = h @ Wq/Wk
    gemm(HBF, wq, QBF, nullptr, nullptr, BT, CC, CC, CC, CC, CC,
         0, 0, 0, 0, 0, 0, 1, 1, 1.0f, GF_OUTBF16);
    gemm(HBF, wk, KBF, nullptr, nullptr, BT, CC, CC, CC, CC, CC,
         0, 0, 0, 0, 0, 0, 1, 1, 1.0f, GF_OUTBF16);
    // v stored transposed per batch: VTB[b][n][t], ldc = T
    gemm(HBF, wv, VTB, nullptr, nullptr, TT, CC, CC, CC, CC, TT,
         (long)TT * CC, 0, 0, 0, (long)CC * TT, 0, BBA, 1, 1.0f,
         GF_OUTBF16 | GF_TRANS);

    // scores[b,h] = q_bh @ k_bh^T * (1/8) : M=N=1024, K=64, batched over (b,h)
    gemm(QBF, KBF, SC, nullptr, nullptr, TT, TT, DHD, CC, CC, TT,
         (long)TT * CC, (long)DHD, (long)TT * CC, (long)DHD,
         (long)HHD * TT * TT, (long)TT * TT, BBA, HHD, 0.125f, 0);

    softmax_causal_kernel<<<BBA * HHD * TT, 256, 0, stream>>>(SC, PR);

    // o[b,h] = P_bh @ v_bh : M=1024, N=64, K=1024 (Bt = VTB slice, K contiguous)
    gemm(PR, VTB, OBF, nullptr, nullptr, TT, DHD, TT, TT, TT, CC,
         (long)HHD * TT * TT, (long)TT * TT, (long)CC * TT, (long)DHD * TT,
         (long)TT * CC, (long)DHD, BBA, HHD, 1.0f, GF_OUTBF16);

    // x = x + o @ Wproj + bproj
    gemm(OBF, wp, XF, bproj + l * CC, XF, BT, CC, CC, CC, CC, CC,
         0, 0, 0, 0, 0, 0, 1, 1, 1.0f, 0);

    layernorm_kernel<<<BT, 256, 0, stream>>>(XF, ln2_g + l * CC, ln2_b + l * CC, HBF);

    // a1 = relu(h2 @ W1 + b1)
    gemm(HBF, w1, A1, b1 + l * HIDD, nullptr, BT, HIDD, CC, CC, CC, HIDD,
         0, 0, 0, 0, 0, 0, 1, 1, 1.0f, GF_RELU | GF_OUTBF16);
    // x = x + a1 @ W2 + b2
    gemm(A1, w2, XF, b2 + l * CC, XF, BT, CC, HIDD, HIDD, HIDD, CC,
         0, 0, 0, 0, 0, 0, 1, 1, 1.0f, 0);
  }

  // ---- 4) LM head + loss ----
  f32_to_bf16_kernel<<<(BT * CC) / 256, 256, 0, stream>>>(XF, XBF);
  gemm(XBF, WLMT, out, blm, nullptr, BT, VV, CC, CC, CC, VV,
       0, 0, 0, 0, 0, 0, 1, 1, 1.0f, 0);

  row_loss_kernel<<<BT, 256, 0, stream>>>(out, targets, RL);
  loss_reduce_kernel<<<1, 256, 0, stream>>>(RL, out + (size_t)BT * VV);
}
